// Graph_MultiHeadAttention_53747220742568
// MI455X (gfx1250) — matrix operand order repaired and checked by reading, not compiled, not run on hardware
//
#include <hip/hip_runtime.h>

#define DV 128   // d_model
#define HN 8     // heads
#define HD 16    // head dim
#define ASTR (DV + 8)    // A-tile LDS row stride (halfwords): 136 -> 16B-aligned rows
#define BSTR (DV + 2)    // B-tile LDS row stride (halfwords): 130 -> bank-staggered

typedef __attribute__((ext_vector_type(16))) __bf16 v16bf;
typedef __attribute__((ext_vector_type(8)))  float  v8f;

// f32 -> bf16 round-to-nearest-even, as raw u16
static __device__ __forceinline__ unsigned short f2bf_u(float f) {
  unsigned int u = __float_as_uint(f);
  u += 0x7fffu + ((u >> 16) & 1u);
  return (unsigned short)(u >> 16);
}
// pack two f32 -> one u32 holding two bf16 (lo, hi)
static __device__ __forceinline__ unsigned int f2bf_pk(float lo, float hi) {
  return (unsigned int)f2bf_u(lo) | ((unsigned int)f2bf_u(hi) << 16);
}

union Frag16 { v16bf v; unsigned int u32[8]; unsigned short u16[16]; };

// ---------------------------------------------------------------------------
// Pre-pass 1: node f32 -> bf16 row-major (packed stores, coalesced)
// ---------------------------------------------------------------------------
__global__ void __launch_bounds__(256)
convert_node_kernel(const float* __restrict__ node, unsigned int* __restrict__ node_bf,
                    long long n4 /* = N*DV/4 */) {
  long long i = (long long)blockIdx.x * 256 + threadIdx.x;
  if (i >= n4) return;
  float4 x = ((const float4*)node)[i];
  uint2 p;
  p.x = f2bf_pk(x.x, x.y);
  p.y = f2bf_pk(x.z, x.w);
  ((uint2*)node_bf)[i] = p;
}

// ---------------------------------------------------------------------------
// Pre-pass 2: five 128x128 weights f32 row-major -> bf16 TRANSPOSED (col-major)
// wbf[y][n*128 + k] = bf16(W[k*128 + n])
// ---------------------------------------------------------------------------
__global__ void __launch_bounds__(256)
convert_w_kernel(const float* __restrict__ W0, const float* __restrict__ W1,
                 const float* __restrict__ W2, const float* __restrict__ W3,
                 const float* __restrict__ W4, unsigned short* __restrict__ wbf) {
  const float* Ws[5] = {W0, W1, W2, W3, W4};
  const float* src = Ws[blockIdx.y];
  unsigned short* dst = wbf + (size_t)blockIdx.y * DV * DV;
  int idx = blockIdx.x * 256 + threadIdx.x;    // 0 .. 16383
  int k = idx >> 7, n = idx & 127;
  dst[(size_t)n * DV + k] = f2bf_u(src[(size_t)k * DV + n]);
}

// ---------------------------------------------------------------------------
// Init: segmax = -inf, denom = 0, agg = 0
// ---------------------------------------------------------------------------
__global__ void __launch_bounds__(256)
init_ws_kernel(float* __restrict__ segmax, float* __restrict__ denom,
               float* __restrict__ agg, int N) {
  int idx = blockIdx.x * 256 + threadIdx.x;
  if (idx < N * HN) { segmax[idx] = -__builtin_inff(); denom[idx] = 0.0f; }
  if (idx < N * DV) agg[idx] = 0.0f;
}

// ---------------------------------------------------------------------------
// Shared WMMA 16x16 tile over K=128: A from LDS (row-major bf16 tile),
// B from LDS (col-major bf16, row = output column). Contiguous k = 16*half + s in both A and B.
// ---------------------------------------------------------------------------
static __device__ __forceinline__ v8f
wmma_k128(const unsigned short* __restrict__ AsLane,   // &As[l16][0]
          const unsigned short* __restrict__ BsLane,   // &Bs[(n0+l16)*BSTR]
          int half, v8f acc) {
  const unsigned int* arow = (const unsigned int*)AsLane;
  const unsigned int* brow = (const unsigned int*)BsLane;
  #pragma unroll
  for (int kk = 0; kk < DV; kk += 32) {
    Frag16 a, b;
    #pragma unroll
    for (int p = 0; p < 8; ++p) {   // A: K0 = kk + half*16 + 2*p
      int K0 = kk + half * 16 + (p << 1);
      a.u32[p] = arow[K0 >> 1];
    }
    int kb = kk + half * 16;        // B: lanes 0-15 K 0-15, lanes 16-31 K 16-31
    #pragma unroll
    for (int p = 0; p < 8; ++p) b.u32[p] = brow[(kb >> 1) + p];
    acc = __builtin_amdgcn_wmma_f32_16x16x32_bf16(false, a.v, false, b.v,
                                                  (short)0, acc, false, false);
  }
  return acc;
}

// cooperative LDS staging of the full 128x128 bf16 weight (col-major src)
static __device__ __forceinline__ void
stage_B(unsigned short* __restrict__ Bs, const unsigned short* __restrict__ Wt) {
  #pragma unroll
  for (int i = 0; i < 8; ++i) {
    int c = threadIdx.x + i * 256;      // 2048 uint4 chunks
    int n = c >> 4, j = c & 15;         // row n, 8-halfword chunk j
    uint4 x = ((const uint4*)(Wt + (size_t)n * DV))[j];
    unsigned int* d = (unsigned int*)(Bs + (size_t)n * BSTR) + j * 4;
    d[0] = x.x; d[1] = x.y; d[2] = x.z; d[3] = x.w;
  }
}

// ---------------------------------------------------------------------------
// Fused q/k/v/skip projection (bf16 inputs preconverted), blockIdx.y selects
// matrix. One wave = one 16x16 output tile; 4x v_wmma_f32_16x16x32_bf16.
// ---------------------------------------------------------------------------
__global__ void __launch_bounds__(256)
gemm4_bf16_kernel(const unsigned short* __restrict__ node_bf,
                  const unsigned short* __restrict__ wbf,
                  const float* __restrict__ b0, const float* __restrict__ b1,
                  const float* __restrict__ b2, const float* __restrict__ b3,
                  float* __restrict__ o0, float* __restrict__ o1,
                  float* __restrict__ o2, float* __restrict__ o3, int N) {
  __shared__ __align__(16) unsigned short As[16][ASTR];
  __shared__ __align__(16) unsigned short Bs[DV * BSTR];
  const int row0 = blockIdx.x * 16;
  const float* bm; float* om;
  switch (blockIdx.y) {
    case 0:  bm = b0; om = o0; break;
    case 1:  bm = b1; om = o1; break;
    case 2:  bm = b2; om = o2; break;
    default: bm = b3; om = o3; break;
  }
  const unsigned short* Wt = wbf + (size_t)blockIdx.y * DV * DV;

  // A tile: 16 rows x 128 bf16, one b128 per thread (coalesced)
  {
    int r = threadIdx.x >> 4, c = threadIdx.x & 15;
    uint4 x = make_uint4(0u, 0u, 0u, 0u);
    if (row0 + r < N) x = ((const uint4*)(node_bf + (size_t)(row0 + r) * DV))[c];
    ((uint4*)&As[r][0])[c] = x;
  }
  stage_B(&Bs[0], Wt);
  __syncthreads();

  const int lane = threadIdx.x & 31;
  const int half = lane >> 4;
  const int l16  = lane & 15;
  const int n0   = (threadIdx.x >> 5) * 16;

  v8f acc;
  float bcol = bm[n0 + l16];
  #pragma unroll
  for (int r = 0; r < 8; ++r) acc[r] = bcol;

  acc = wmma_k128(&As[l16][0], &Bs[(size_t)(n0 + l16) * BSTR], half, acc);

  #pragma unroll
  for (int r = 0; r < 8; ++r) {      // C/D: row = r + 8*half, col = l16
    int row = row0 + r + half * 8;
    if (row < N) om[(size_t)row * DV + n0 + l16] = acc[r];
  }
}

// ---------------------------------------------------------------------------
// Edge scores: alpha[e,h] = dot(q[dst,h,:], k[src,h,:]) / 4; atomic seg max
// ---------------------------------------------------------------------------
__global__ void __launch_bounds__(256)
edge_scores_kernel(const int* __restrict__ esrc, const int* __restrict__ edst,
                   const float* __restrict__ q, const float* __restrict__ k,
                   float* __restrict__ alpha, float* __restrict__ segmax, int E) {
  int idx = blockIdx.x * 256 + threadIdx.x;
  if (idx >= E * HN) return;
  int e = idx >> 3, h = idx & 7;
  int s = esrc[e], d = edst[e];
  const float4* qp = (const float4*)(q + (size_t)d * DV + h * HD);
  const float4* kp = (const float4*)(k + (size_t)s * DV + h * HD);
  float acc = 0.0f;
  #pragma unroll
  for (int i = 0; i < 4; ++i) {
    float4 a = qp[i], b = kp[i];
    acc += a.x * b.x + a.y * b.y + a.z * b.z + a.w * b.w;
  }
  acc *= 0.25f;                            // 1/sqrt(16)
  alpha[idx] = acc;
  float* mp = segmax + (size_t)d * HN + h; // float atomic max via int ordering trick
  if (acc >= 0.0f) atomicMax((int*)mp, __float_as_int(acc));
  else             atomicMin((unsigned int*)mp, __float_as_uint(acc));
}

// ---------------------------------------------------------------------------
// ex = exp(alpha - segmax[dst]); denom[dst] += ex  (alpha overwritten by ex)
// ---------------------------------------------------------------------------
__global__ void __launch_bounds__(256)
edge_expsum_kernel(const int* __restrict__ edst, const float* __restrict__ segmax,
                   float* __restrict__ alpha, float* __restrict__ denom, int E) {
  int idx = blockIdx.x * 256 + threadIdx.x;
  if (idx >= E * HN) return;
  int e = idx >> 3, h = idx & 7;
  int d = edst[e];
  float ex = __expf(alpha[idx] - segmax[(size_t)d * HN + h]);
  alpha[idx] = ex;
  atomicAdd(denom + (size_t)d * HN + h, ex);
}

// ---------------------------------------------------------------------------
// agg[dst, :] += v[src, :] * (ex / denom[dst]); one thread per (edge, channel)
// ---------------------------------------------------------------------------
__global__ void __launch_bounds__(256)
edge_scatter_kernel(const int* __restrict__ esrc, const int* __restrict__ edst,
                    const float* __restrict__ v, const float* __restrict__ alpha,
                    const float* __restrict__ denom, float* __restrict__ agg, int E) {
  long long idx = (long long)blockIdx.x * 256 + threadIdx.x;
  if (idx >= (long long)E * DV) return;
  int e = (int)(idx >> 7), c = (int)(idx & 127);
  int h = c >> 4;
  int s = esrc[e], d = edst[e];
  float w = alpha[(size_t)e * HN + h] / denom[(size_t)d * HN + h];
  atomicAdd(agg + (size_t)d * DV + c, v[(size_t)s * DV + c] * w);
}

// ---------------------------------------------------------------------------
// Final: out = (agg + skip) @ Wout + bout  (A add+convert fused into tile load)
// ---------------------------------------------------------------------------
__global__ void __launch_bounds__(256)
gemm_out_bf16_kernel(const float* __restrict__ agg, const float* __restrict__ skip,
                     const unsigned short* __restrict__ Wt, const float* __restrict__ b,
                     float* __restrict__ out, int N) {
  __shared__ __align__(16) unsigned short As[16][ASTR];
  __shared__ __align__(16) unsigned short Bs[DV * BSTR];
  const int row0 = blockIdx.x * 16;

  for (int i = threadIdx.x; i < 16 * (DV / 2); i += 256) {   // packed pair stores
    int r = i >> 6, c2 = (i & 63) * 2;
    unsigned int pk = 0u;
    if (row0 + r < N) {
      size_t off = (size_t)(row0 + r) * DV + c2;
      pk = f2bf_pk(agg[off] + skip[off], agg[off + 1] + skip[off + 1]);
    }
    *(unsigned int*)&As[r][c2] = pk;
  }
  stage_B(&Bs[0], Wt);
  __syncthreads();

  const int lane = threadIdx.x & 31;
  const int half = lane >> 4;
  const int l16  = lane & 15;
  const int n0   = (threadIdx.x >> 5) * 16;

  v8f acc;
  float bcol = b[n0 + l16];
  #pragma unroll
  for (int r = 0; r < 8; ++r) acc[r] = bcol;

  acc = wmma_k128(&As[l16][0], &Bs[(size_t)(n0 + l16) * BSTR], half, acc);

  #pragma unroll
  for (int r = 0; r < 8; ++r) {
    int row = row0 + r + half * 8;
    if (row < N) out[(size_t)row * DV + n0 + l16] = acc[r];
  }
}

// ---------------------------------------------------------------------------
extern "C" void kernel_launch(void* const* d_in, const int* in_sizes, int n_in,
                              void* d_out, int out_size, void* d_ws, size_t ws_size,
                              hipStream_t stream) {
  const float* node = (const float*)d_in[0];
  const int*   ei   = (const int*)d_in[1];
  const float* Wq = (const float*)d_in[2];  const float* bq = (const float*)d_in[3];
  const float* Wk = (const float*)d_in[4];  const float* bk = (const float*)d_in[5];
  const float* Wv = (const float*)d_in[6];  const float* bv = (const float*)d_in[7];
  const float* Ws = (const float*)d_in[8];  const float* bs = (const float*)d_in[9];
  const float* Wo = (const float*)d_in[10]; const float* bo = (const float*)d_in[11];
  float* out = (float*)d_out;

  const int N = in_sizes[0] / DV;   // 50000
  const int E = in_sizes[1] / 2;    // 800000
  const int* esrc = ei;
  const int* edst = ei + E;

  char* ws = (char*)d_ws;
  float* q    = (float*)ws; ws += (size_t)N * DV * sizeof(float);
  float* kb_  = (float*)ws; ws += (size_t)N * DV * sizeof(float);
  float* vb_  = (float*)ws; ws += (size_t)N * DV * sizeof(float);
  float* skip = (float*)ws; ws += (size_t)N * DV * sizeof(float);
  float* agg  = (float*)ws; ws += (size_t)N * DV * sizeof(float);
  float* alph = (float*)ws; ws += (size_t)E * HN * sizeof(float);
  float* segm = (float*)ws; ws += (size_t)N * HN * sizeof(float);
  float* den  = (float*)ws; ws += (size_t)N * HN * sizeof(float);
  unsigned short* node_bf = (unsigned short*)ws; ws += (size_t)N * DV * sizeof(unsigned short);
  unsigned short* wbf     = (unsigned short*)ws; // 5 * 128*128 bf16

  dim3 blk(256);

  // pre-convert: node -> bf16, weights -> bf16 transposed
  long long n4 = (long long)N * DV / 4;
  convert_node_kernel<<<(unsigned)((n4 + 255) / 256), blk, 0, stream>>>(
      node, (unsigned int*)node_bf, n4);
  convert_w_kernel<<<dim3(DV * DV / 256, 5), blk, 0, stream>>>(Wq, Wk, Wv, Ws, Wo, wbf);

  init_ws_kernel<<<((size_t)N * DV + 255) / 256, blk, 0, stream>>>(segm, den, agg, N);

  dim3 g1((N + 15) / 16, 4);
  gemm4_bf16_kernel<<<g1, blk, 0, stream>>>(node_bf, wbf, bq, bk, bv, bs,
                                            q, kb_, vb_, skip, N);

  int eh = E * HN;
  edge_scores_kernel<<<(eh + 255) / 256, blk, 0, stream>>>(esrc, edst, q, kb_, alph, segm, E);
  edge_expsum_kernel<<<(eh + 255) / 256, blk, 0, stream>>>(edst, segm, alph, den, E);

  long long ed = (long long)E * DV;
  edge_scatter_kernel<<<(unsigned)((ed + 255) / 256), blk, 0, stream>>>(esrc, edst, vb_,
                                                                       alph, den, agg, E);

  gemm_out_bf16_kernel<<<dim3((N + 15) / 16, 1), blk, 0, stream>>>(
      agg, skip, wbf + 4 * DV * DV, bo, out, N);
}
